// EELayer_77455440216524
// MI455X (gfx1250) — matrix operand order repaired and checked by reading, not compiled, not run on hardware
//
#include <hip/hip_runtime.h>
#include <hip/hip_bf16.h>

typedef __attribute__((ext_vector_type(16))) _Float16 v16h;
typedef __attribute__((ext_vector_type(8)))  _Float16 v8h;
typedef __attribute__((ext_vector_type(8)))  float    v8f;

#define NN 50000
#define DEG 16
#define NE (NN * DEG)

// ---------------------------------------------------------------------------
// WMMA fragment loaders (wave32, V_WMMA_F32_16X16X32_F16 layouts)
// A (16x32 f16): lane m=lane&15 holds row m; khi=(lane>>4)*8;
//   halves 0..7 -> K = kb+khi+0..7 ; halves 8..15 -> K = kb+khi+16..23
// B (32x16 f16): lane col n=lane&15; klo=kb+(lane>>4)*8; halves 0..7 -> K=klo+0..7,
//   halves 8..15 -> K=klo+16..23 (two 16B loads from transposed WT[n][k]).
// C/D (16x16 f32, 8 VGPRs): reg r, lane l -> M = r + 8*(l>>4), Ncol = l&15.
// ---------------------------------------------------------------------------
__device__ __forceinline__ v16h a_from_f16(const _Float16* X, int ld, int kb, int lane) {
  const int m   = lane & 15;
  const int khi = (lane >> 4) << 3;
  const _Float16* p = X + m * ld + kb + khi;
  union { v16h v; v8h h[2]; } u;
  u.h[0] = *(const v8h*)(p);
  u.h[1] = *(const v8h*)(p + 16);
  return u.v;
}

__device__ __forceinline__ v16h a_from_f32(const float* X, int ld, int kb, int lane) {
  const int m   = lane & 15;
  const int khi = (lane >> 4) << 3;
  const float* p = X + m * ld + kb + khi;
  v16h a;
#pragma unroll
  for (int i = 0; i < 8; ++i) a[i] = (_Float16)p[i];
#pragma unroll
  for (int i = 0; i < 8; ++i) a[8 + i] = (_Float16)p[16 + i];
  return a;
}

__device__ __forceinline__ v16h b_frag(const _Float16* WT, int K, int n0, int kb, int lane) {
  const int n   = n0 + (lane & 15);
  const int klo = kb + ((lane >> 4) << 3);
  const _Float16* p = WT + (size_t)n * K + klo;
  union { v16h v; v8h h[2]; } u;
  u.h[0] = *(const v8h*)(p);
  u.h[1] = *(const v8h*)(p + 16);
  return u.v;
}

__device__ __forceinline__ v8f wmma_acc(v16h a, v16h b, v8f c) {
  return __builtin_amdgcn_wmma_f32_16x16x32_f16(false, a, false, b, (short)0, c, false, false);
}

__device__ __forceinline__ void wmma4(v16h a, const _Float16* WT, int K, int kb, int lane, v8f* acc) {
#pragma unroll
  for (int nt = 0; nt < 4; ++nt) {
    v16h b = b_frag(WT, K, nt * 16, kb, lane);
    acc[nt] = wmma_acc(a, b, acc[nt]);
  }
}

__device__ __forceinline__ void acc_zero(v8f* acc) {
  const v8f z = {0.f, 0.f, 0.f, 0.f, 0.f, 0.f, 0.f, 0.f};
  acc[0] = z; acc[1] = z; acc[2] = z; acc[3] = z;
}

// GEMM: [16 x K] (f16, row-major, stride ldx) @ W[K x 64] via WT[64][K] f16
__device__ __forceinline__ void gemm_f16(const _Float16* X, int ldx, int K,
                                         const _Float16* WT, int lane, v8f* acc) {
  acc_zero(acc);
  for (int kb = 0; kb < K; kb += 32) {
    v16h a = a_from_f16(X, ldx, kb, lane);
    wmma4(a, WT, K, kb, lane, acc);
  }
}

// GEMM with f32 source rows (K = 64)
__device__ __forceinline__ void gemm_f32src(const float* X, const _Float16* WT, int lane, v8f* acc) {
  acc_zero(acc);
#pragma unroll
  for (int kb = 0; kb < 64; kb += 32) {
    v16h a = a_from_f32(X, 64, kb, lane);
    wmma4(a, WT, 64, kb, lane, acc);
  }
}

// hidden = sin(acc + b1) -> LDS f16 [16][64]
__device__ __forceinline__ void store_hidden_sin(const v8f* acc, const float* b1, _Float16* hW, int lane) {
  const int colb = lane & 15;
  const int mhi  = (lane >> 4) << 3;
#pragma unroll
  for (int nt = 0; nt < 4; ++nt) {
    float bb = b1[nt * 16 + colb];
#pragma unroll
    for (int r = 0; r < 8; ++r)
      hW[(mhi + r) * 64 + nt * 16 + colb] = (_Float16)__sinf(acc[nt][r] + bb);
  }
}

// cooperative f16 copy global -> LDS (16B granules)
__device__ __forceinline__ void copy_w_lds(_Float16* dst, const _Float16* src,
                                           int nhalves, int tid, int nthreads) {
  const int nv = nhalves >> 3;
  for (int i = tid; i < nv; i += nthreads)
    ((v8h*)dst)[i] = ((const v8h*)src)[i];
}

// ---------------------------------------------------------------------------
// Weight pack: W[K][Nc] f32 -> WT[Nc][K] f16
// ---------------------------------------------------------------------------
__global__ void packWT_kernel(const float* __restrict__ W, _Float16* __restrict__ WT, int K, int Nc) {
  int i = blockIdx.x * 256 + threadIdx.x;
  if (i >= K * Nc) return;
  int k = i / Nc, n = i - k * Nc;
  WT[(size_t)n * K + k] = (_Float16)W[(size_t)k * Nc + n];
}

// ---------------------------------------------------------------------------
// Y = mlp(X) = sin(X@W1+b1)@W2+b2 over rows of 64 feats (ftnn / ften)
// Weights staged in LDS once per block; grid-stride over tiles.
// A memory-clobber per iteration stops LICM from hoisting (and then spilling)
// the loop-invariant LDS weight fragments.
// ---------------------------------------------------------------------------
__global__ void mlp64_kernel(const float* __restrict__ X,
                             const _Float16* __restrict__ W1T, const float* __restrict__ b1,
                             const _Float16* __restrict__ W2T, const float* __restrict__ b2,
                             float* __restrict__ Y, int ntiles) {
  __shared__ _Float16 w1s[64 * 64];
  __shared__ _Float16 w2s[64 * 64];
  __shared__ _Float16 hAll[8][16 * 64];
  const int lane = threadIdx.x & 31;
  const int wid  = threadIdx.x >> 5;
  const int gw   = blockIdx.x * 8 + wid;
  const int nw   = gridDim.x * 8;
  _Float16* hW = hAll[wid];
  const int colb = lane & 15, mhi = (lane >> 4) << 3;

  copy_w_lds(w1s, W1T, 4096, threadIdx.x, 256);
  copy_w_lds(w2s, W2T, 4096, threadIdx.x, 256);
  __syncthreads();

  for (int tile = gw; tile < ntiles; tile += nw) {
    asm volatile("" ::: "memory");  // defeat LICM of LDS weight fragments
    const float* Xt = X + (size_t)tile * 16 * 64;
    if (tile + nw < ntiles)
      __builtin_prefetch(X + (size_t)(tile + nw) * 16 * 64 + lane * 32, 0, 0);

    v8f acc[4];
    gemm_f32src(Xt, w1s, lane, acc);
    store_hidden_sin(acc, b1, hW, lane);
    gemm_f16(hW, 64, 64, w2s, lane, acc);
#pragma unroll
    for (int nt = 0; nt < 4; ++nt) {
      float bb = b2[nt * 16 + colb];
#pragma unroll
      for (int r = 0; r < 8; ++r)
        Y[((size_t)tile * 16 + mhi + r) * 64 + nt * 16 + colb] = acc[nt][r] + bb;
    }
  }
}

// ---------------------------------------------------------------------------
// Node update: softmax attention stats + ptn + fused ftne
// dst = repeat(arange(N),16) -> edges of node n are rows 16n..16n+15
// ---------------------------------------------------------------------------
#define K3_WAVE_BYTES 17408  // xW 10240 + hvW 4096 + aW 1024 + hW 2048

__global__ void node_update_kernel(const float* __restrict__ hvG, const float* __restrict__ heG,
                                   const int* __restrict__ srcG,
                                   const _Float16* ptnW1T, const float* ptnB1,
                                   const _Float16* ptnW2T, const float* ptnB2,
                                   const _Float16* ftneW1T, const float* ftneB1,
                                   const _Float16* ftneW2T, const float* ftneB2,
                                   float* __restrict__ outNode, _Float16* __restrict__ hv2G,
                                   int ntiles) {
  extern __shared__ unsigned char smemRaw[];
  const int lane = threadIdx.x & 31;
  const int wid  = threadIdx.x >> 5;
  unsigned char* wb = smemRaw + (size_t)wid * K3_WAVE_BYTES;
  _Float16* xW  = (_Float16*)wb;            // [16][320] f16 : [mean|min|max|std|hv]
  float*    hvW = (float*)(wb + 10240);     // [16][64]
  float*    aW  = (float*)(wb + 14336);     // [16][16] logits -> attn
  _Float16* hW  = (_Float16*)(wb + 15360);  // [16][64] hidden
  int tile = blockIdx.x * 8 + wid;
  if (tile >= ntiles) tile = ntiles - 1;
  const int nodeBase = tile * 16;

  // hv rows -> LDS (f32 for dots) and x cols 256..319 (f16)
  for (int idx = lane; idx < 1024; idx += 32) {
    int r = idx >> 6, f = idx & 63;
    float v = hvG[(size_t)(nodeBase + r) * 64 + f];
    hvW[idx] = v;
    xW[r * 320 + 256 + f] = (_Float16)v;
  }

  // logits: dot(hv[src], hv[dst])
  for (int t = lane; t < 256; t += 32) {
    int r = t >> 4, j = t & 15;
    int s = srcG[(nodeBase + r) * 16 + j];
    const float* hs = hvG + (size_t)s * 64;
    const float* hd = hvW + r * 64;
    float dot = 0.f;
#pragma unroll 8
    for (int f = 0; f < 64; ++f) dot += hd[f] * hs[f];
    aW[t] = dot;
  }

  // segment softmax / sqrt(DE)  (16 edges per node, lanes 0..15)
  if (lane < 16) {
    const int r = lane;
    float m = -3.4e38f;
#pragma unroll
    for (int j = 0; j < 16; ++j) m = fmaxf(m, aW[r * 16 + j]);
    float ex[16]; float z = 0.f;
#pragma unroll
    for (int j = 0; j < 16; ++j) { ex[j] = __expf(aW[r * 16 + j] - m); z += ex[j]; }
    float inv = 1.0f / (z * 8.0f);  // /sqrt(64)
#pragma unroll
    for (int j = 0; j < 16; ++j) aW[r * 16 + j] = ex[j] * inv;
  }

  // weighted mean/min/max/std over he -> x cols 0..255
  for (int t = lane; t < 1024; t += 32) {
    int r = t >> 6, f = t & 63;
    const float* he = heG + ((size_t)(nodeBase + r) * 16) * 64 + f;
    float mean = 0.f, sq = 0.f, mn = 3.4e38f, mx = -3.4e38f;
#pragma unroll
    for (int j = 0; j < 16; ++j) {
      float a  = aW[r * 16 + j];
      float h  = he[j * 64];
      float ah = a * h;
      mean += ah; sq += ah * h;
      mn = fminf(mn, ah); mx = fmaxf(mx, ah);
    }
    float sd = sqrtf(fmaxf(sq - mean * mean, 0.f) + 1e-5f);
    xW[r * 320 + f]       = (_Float16)mean;
    xW[r * 320 + 64 + f]  = (_Float16)mn;
    xW[r * 320 + 128 + f] = (_Float16)mx;
    xW[r * 320 + 192 + f] = (_Float16)sd;
  }

  const int colb = lane & 15, mhi = (lane >> 4) << 3;
  v8f acc[4];

  // ptn: [16,320] @ [320,64] -> sin -> [16,64] @ [64,64] -> sin
  gemm_f16(xW, 320, 320, ptnW1T, lane, acc);
  store_hidden_sin(acc, ptnB1, hW, lane);
  gemm_f16(hW, 64, 64, ptnW2T, lane, acc);

  _Float16* nnW = xW;  // reuse as [16][64] node_new (x no longer needed)
#pragma unroll
  for (int nt = 0; nt < 4; ++nt) {
    float bb = ptnB2[nt * 16 + colb];
#pragma unroll
    for (int r = 0; r < 8; ++r) {
      float v = __sinf(acc[nt][r] + bb);
      int m = mhi + r, c = nt * 16 + colb;
      outNode[(size_t)(nodeBase + m) * 64 + c] = v;
      nnW[m * 64 + c] = (_Float16)v;
    }
  }

  // fused ftne: hv2 = mlp(node_new)  (no outer sin)
  gemm_f16(nnW, 64, 64, ftneW1T, lane, acc);
  store_hidden_sin(acc, ftneB1, hW, lane);
  gemm_f16(hW, 64, 64, ftneW2T, lane, acc);
#pragma unroll
  for (int nt = 0; nt < 4; ++nt) {
    float bb = ftneB2[nt * 16 + colb];
#pragma unroll
    for (int r = 0; r < 8; ++r)
      hv2G[(size_t)(nodeBase + mhi + r) * 64 + nt * 16 + colb] = (_Float16)(acc[nt][r] + bb);
  }
}

// ---------------------------------------------------------------------------
// Edge update: he2 = fte(edge_feats) recomputed in-kernel; x = [hv2[d]+hv2[s], he2]
// edge_new = sin(mlp_pte(x)); all 4 weight matrices staged in LDS (40KB).
// Dynamic LDS layout:
//   [0,8K)    fteW1T   [8K,16K)  fteW2T   [16K,32K) pteW1T   [32K,40K) pteW2T
//   [40K + wid*6K): per-wave xW [16][128] f16 (4KB) + hW [16][64] f16 (2KB)
// ---------------------------------------------------------------------------
#define K4_LDS_BYTES (40960 + 8 * 6144)

__global__ void edge_update_kernel(const float* __restrict__ edgeF,
                                   const int* __restrict__ srcG, const int* __restrict__ dstG,
                                   const _Float16* __restrict__ hv2G,
                                   const _Float16* fteW1T, const float* fteB1,
                                   const _Float16* fteW2T, const float* fteB2,
                                   const _Float16* pteW1T, const float* pteB1,
                                   const _Float16* pteW2T, const float* pteB2,
                                   float* __restrict__ outEdge, int ntiles) {
  extern __shared__ unsigned char esm[];
  _Float16* fte1s = (_Float16*)(esm);
  _Float16* fte2s = (_Float16*)(esm + 8192);
  _Float16* pte1s = (_Float16*)(esm + 16384);
  _Float16* pte2s = (_Float16*)(esm + 32768);
  const int lane = threadIdx.x & 31;
  const int wid  = threadIdx.x >> 5;
  const int gw   = blockIdx.x * 8 + wid;
  const int nw   = gridDim.x * 8;
  unsigned char* wvb = esm + 40960 + (size_t)wid * 6144;
  _Float16* xW = (_Float16*)wvb;
  _Float16* hW = (_Float16*)(wvb + 4096);
  const int colb = lane & 15, mhi = (lane >> 4) << 3;

  copy_w_lds(fte1s, fteW1T, 4096, threadIdx.x, 256);
  copy_w_lds(fte2s, fteW2T, 4096, threadIdx.x, 256);
  copy_w_lds(pte1s, pteW1T, 8192, threadIdx.x, 256);
  copy_w_lds(pte2s, pteW2T, 4096, threadIdx.x, 256);
  __syncthreads();

  for (int tile = gw; tile < ntiles; tile += nw) {
    asm volatile("" ::: "memory");  // defeat LICM of LDS weight fragments
    const int eBase = tile * 16;
    const float* Xe = edgeF + (size_t)eBase * 64;
    if (tile + nw < ntiles)
      __builtin_prefetch(edgeF + (size_t)(tile + nw) * 16 * 64 + lane * 32, 0, 0);

    v8f acc[4];

    // he2 = mlp_fte(edge_feats) -> x cols 64..127
    gemm_f32src(Xe, fte1s, lane, acc);
    store_hidden_sin(acc, fteB1, hW, lane);
    gemm_f16(hW, 64, 64, fte2s, lane, acc);
#pragma unroll
    for (int nt = 0; nt < 4; ++nt) {
      float bb = fteB2[nt * 16 + colb];
#pragma unroll
      for (int r = 0; r < 8; ++r)
        xW[(mhi + r) * 128 + 64 + nt * 16 + colb] = (_Float16)(acc[nt][r] + bb);
    }

    // hv2[dst] + hv2[src] -> x cols 0..63
    for (int t = lane; t < 1024; t += 32) {
      int r = t >> 6, f = t & 63;
      int e = eBase + r;
      int s = srcG[e], d = dstG[e];
      float v = (float)hv2G[(size_t)s * 64 + f] + (float)hv2G[(size_t)d * 64 + f];
      xW[r * 128 + f] = (_Float16)v;
    }

    // pte: [16,128] @ [128,64] -> sin -> [16,64] @ [64,64] -> sin
    gemm_f16(xW, 128, 128, pte1s, lane, acc);
    store_hidden_sin(acc, pteB1, hW, lane);
    gemm_f16(hW, 64, 64, pte2s, lane, acc);
#pragma unroll
    for (int nt = 0; nt < 4; ++nt) {
      float bb = pteB2[nt * 16 + colb];
#pragma unroll
      for (int r = 0; r < 8; ++r)
        outEdge[(size_t)(eBase + mhi + r) * 64 + nt * 16 + colb] = __sinf(acc[nt][r] + bb);
    }
  }
}

// ---------------------------------------------------------------------------
extern "C" void kernel_launch(void* const* d_in, const int* in_sizes, int n_in,
                              void* d_out, int out_size, void* d_ws, size_t ws_size,
                              hipStream_t stream) {
  // setup_inputs() order: 6 MLPs x (W1,b1,W2,b2), then node_feats, edge_feats, src, dst
  const float* W[24];
  for (int i = 0; i < 24; ++i) W[i] = (const float*)d_in[i];
  const float* nodeF = (const float*)d_in[24];
  const float* edgeF = (const float*)d_in[25];
  const int*   src   = (const int*)d_in[26];
  const int*   dst   = (const int*)d_in[27];
  float* outNode = (float*)d_out;
  float* outEdge = outNode + (size_t)NN * 64;

  char* ws = (char*)d_ws;
  size_t off = 0;
  auto take = [&](size_t bytes) { size_t o = off; off += (bytes + 255) & ~(size_t)255; return o; };
  float*    hv  = (float*)(ws + take((size_t)NN * 64 * 4));
  float*    he  = (float*)(ws + take((size_t)NE * 64 * 4));
  _Float16* hv2 = (_Float16*)(ws + take((size_t)NN * 64 * 2));

  // packed f16 transposed weights
  _Float16* WT[12];
  const int Kdim[12] = {64, 64, 64, 64, 320, 64, 64, 64, 64, 64, 128, 64};
  for (int i = 0; i < 12; ++i) WT[i] = (_Float16*)(ws + take((size_t)Kdim[i] * 64 * 2));
  // WT index: 0:ftnnW1 1:ftnnW2 2:ftenW1 3:ftenW2 4:ptnW1 5:ptnW2
  //           6:ftneW1 7:ftneW2 8:fteW1 9:fteW2 10:pteW1 11:pteW2
  const float* Wsrc[12] = {W[0], W[2], W[4], W[6], W[8], W[10], W[12], W[14], W[16], W[18], W[20], W[22]};
  for (int i = 0; i < 12; ++i) {
    int tot = Kdim[i] * 64;
    packWT_kernel<<<(tot + 255) / 256, 256, 0, stream>>>(Wsrc[i], WT[i], Kdim[i], 64);
  }

  const int nTilesN = NN / 16;  // 3125
  const int nTilesE = NE / 16;  // 50000
  const int blkN_mlp = (nTilesN + 15) / 16;       // ~2 tiles/wave
  const int blkE_mlp = (nTilesE + 31) / 32;       // ~4 tiles/wave
  const int blkN     = (nTilesN + 7) / 8;
  const int blkE_upd = (nTilesE + 31) / 32;       // ~4 tiles/wave

  // hv = mlp_ftnn(node_feats) ; he = mlp_ften(edge_feats)
  mlp64_kernel<<<blkN_mlp, 256, 0, stream>>>(nodeF, WT[0], W[1], WT[1], W[3], hv, nTilesN);
  mlp64_kernel<<<blkE_mlp, 256, 0, stream>>>(edgeF, WT[2], W[5], WT[3], W[7], he, nTilesE);

  // node update (attention stats + ptn + ftne fused)
  node_update_kernel<<<blkN, 256, 8 * K3_WAVE_BYTES, stream>>>(
      hv, he, src,
      WT[4], W[9], WT[5], W[11],
      WT[6], W[13], WT[7], W[15],
      outNode, hv2, nTilesN);

  // edge update (fte fused + pte, weights in LDS)
  edge_update_kernel<<<blkE_upd, 256, K4_LDS_BYTES, stream>>>(
      edgeF, src, dst, hv2,
      WT[8], W[17], WT[9], W[19],
      WT[10], W[21], WT[11], W[23],
      outEdge, nTilesE);
}
